// MultiHeadSlidingAtt_35038343201440
// MI455X (gfx1250) — matrix operand order repaired and checked by reading, not compiled, not run on hardware
//
#include <hip/hip_runtime.h>
#include <hip/hip_bf16.h>

typedef __attribute__((ext_vector_type(16))) _Float16 v16h;
typedef __attribute__((ext_vector_type(8)))  float    v8f;

#define B_    32
#define TQ    64
#define TK    8192
#define DM    512
#define H_    8
#define DK_   64
#define WIN_  97
#define WINP  128   // window padded to K-chunk multiple

union AF { v16h h; uint4 u[2]; };

static __device__ __forceinline__ v8f wmma16(const AF& a, const AF& b, v8f c) {
  // 8 args: (neg_a, A, neg_b, B, c_mod, C, reuse_a, reuse_b)
  return __builtin_amdgcn_wmma_f32_16x16x32_f16(false, a.h, false, b.h,
                                                (short)0, c, false, false);
}

static __device__ __forceinline__ void async_cp16(uint32_t lds_addr, const void* gaddr) {
  uint64_t g = (uint64_t)(uintptr_t)gaddr;
  asm volatile("global_load_async_to_lds_b128 %0, %1, off" :: "v"(lds_addr), "v"(g) : "memory");
}

// ---------------------------------------------------------------------------
// 16x64 output strip over K=512. B-tiles staged through LDS by async DMA
// (double buffered), A fragments double-buffered in registers.
// Block = 256 threads (8 waves) sharing one 64x32 B tile per chunk.
// ---------------------------------------------------------------------------
__device__ __forceinline__ void gemm_strip(const _Float16* __restrict__ X, int rowBase,
                                           const _Float16* __restrict__ Wtile, int ldW,
                                           _Float16 (*Bs)[64 * 32], v8f acc[4]) {
  const int tid  = threadIdx.x;
  const int lane = tid & 31;
  const bool hi  = lane >= 16;
  const int am = lane & 15, akb = hi ? 8 : 0;
  const int bn = lane & 15, bkb = hi ? 8 : 0;
  // cooperative copy mapping: 256 threads x 16B = 4KB chunk (64 rows x 32 halves)
  const int cn = tid >> 2;            // 0..63 : weight row within tile
  const int cs = (tid & 3) * 8;       // 0,8,16,24 : halves within chunk
  const uint32_t lds0 = (uint32_t)(uintptr_t)(&Bs[0][cn * 32 + cs]);
  const uint32_t lds1 = (uint32_t)(uintptr_t)(&Bs[1][cn * 32 + cs]);
  const _Float16* arow = X + (size_t)(rowBase + am) * 512 + akb;

  async_cp16(lds0, Wtile + (size_t)cn * ldW + cs);   // prime buffer 0
  AF a;
  a.u[0] = *(const uint4*)(arow);
  a.u[1] = *(const uint4*)(arow + 16);
  asm volatile("s_wait_asynccnt 0x0" ::: "memory");
  __syncthreads();

  for (int kc = 0; kc < 512; kc += 32) {
    const int buf = (kc >> 5) & 1;
    const bool more = (kc + 32 < 512);
    if (more)   // async-prefetch next weight chunk into the other buffer
      async_cp16(buf ? lds0 : lds1, Wtile + (size_t)cn * ldW + kc + 32 + cs);
    AF an;
    if (more) { // register double-buffer the A fragment
      an.u[0] = *(const uint4*)(arow + kc + 32);
      an.u[1] = *(const uint4*)(arow + kc + 48);
    }
#pragma unroll
    for (int t = 0; t < 4; ++t) {
      AF bf;
      const _Float16* bp = &Bs[buf][(t * 16 + bn) * 32 + bkb];
      bf.u[0] = *(const uint4*)(bp);          // ds_load_b128
      bf.u[1] = *(const uint4*)(bp + 16);
      acc[t] = wmma16(a, bf, acc[t]);
    }
    asm volatile("s_wait_asynccnt 0x0" ::: "memory");
    __syncthreads();
    if (more) a = an;
  }
}

// ---------------- fused f32 -> f16 conversions (query + 4 weight mats) ----------------
__global__ __launch_bounds__(256) void k_prep(const float* __restrict__ q,
                                              const float* __restrict__ Wq,
                                              const float* __restrict__ Wk,
                                              const float* __restrict__ Wv,
                                              const float* __restrict__ Wp,
                                              _Float16* __restrict__ qh,
                                              _Float16* __restrict__ Wqh,
                                              _Float16* __restrict__ Wkh,
                                              _Float16* __restrict__ Wvh,
                                              _Float16* __restrict__ Wph) {
  int i = blockIdx.x * blockDim.x + threadIdx.x;
  const int NQn = B_ * TQ * DM;     // 1048576
  const int NW1 = 512 * 512;        // 262144
  const int NWp = 512 * 1024;       // 524288
  if (i < NQn) { qh[i] = (_Float16)q[i]; return; }
  i -= NQn;
  if (i < NW1) { Wqh[i] = (_Float16)Wq[i]; return; }
  i -= NW1;
  if (i < NW1) { Wkh[i] = (_Float16)Wk[i]; return; }
  i -= NW1;
  if (i < NW1) { Wvh[i] = (_Float16)Wv[i]; return; }
  i -= NW1;
  if (i < NWp) { Wph[i] = (_Float16)Wp[i]; }
}

// ---------------- gather window rows (f32 -> f16, zero pad) ----------------
__global__ __launch_bounds__(256) void k_gather(const float* __restrict__ values,
                                                const int* __restrict__ center,
                                                _Float16* __restrict__ winh) {
  size_t i = (size_t)blockIdx.x * blockDim.x + threadIdx.x;   // B_*WINP*DM
  if (i >= (size_t)B_ * WINP * DM) return;
  int c = (int)(i & (DM - 1));
  size_t row = i >> 9;
  int r = (int)(row & (WINP - 1));
  int b = (int)(row >> 7);
  float v = 0.f;
  if (r < WIN_) {
    int start = center[b] - 48; if (start < 0) start = 0;
    int idx = start + r; if (idx > TK - 1) idx = TK - 1;
    v = values[((size_t)b * TK + idx) * DM + c];
  }
  winh[i] = (_Float16)v;
}

// ---------------- Y(f16, MxN=512) = X @ W^T + bias (Q, K projections) ----------------
__global__ __launch_bounds__(256) void k_gemm_proj(const _Float16* __restrict__ X,
                                                   const _Float16* __restrict__ W,
                                                   const float* __restrict__ bias,
                                                   _Float16* __restrict__ Y) {
  __shared__ alignas(16) _Float16 Bs[2][64 * 32];
  const int lane = threadIdx.x & 31, wave = threadIdx.x >> 5;
  const int n0 = blockIdx.x * 64;
  const int rowBase = blockIdx.y * 128 + wave * 16;
  const bool hi = lane >= 16;
  const int bn = lane & 15;
  v8f acc[4] = {};
  gemm_strip(X, rowBase, W + (size_t)n0 * 512, 512, Bs, acc);
#pragma unroll
  for (int t = 0; t < 4; ++t) {
    int n = n0 + t * 16 + bn;
    float bv = bias[n];
#pragma unroll
    for (int j = 0; j < 8; ++j) {
      int m = rowBase + j + (hi ? 8 : 0);
      Y[(size_t)m * 512 + n] = (_Float16)(acc[t][j] + bv);
    }
  }
}

// ---------------- V projection with transposed epilogue ----------------
// YT[b][n][k] = V[b*WINP + k][n]; D-layout regs j=0..7 are consecutive k,
// so each (lane,tile) emits one packed b128 store.
__global__ __launch_bounds__(256) void k_gemm_projT(const _Float16* __restrict__ X,
                                                    const _Float16* __restrict__ W,
                                                    const float* __restrict__ bias,
                                                    _Float16* __restrict__ YT) {
  __shared__ alignas(16) _Float16 Bs[2][64 * 32];
  const int lane = threadIdx.x & 31, wave = threadIdx.x >> 5;
  const int n0 = blockIdx.x * 64;
  const int rowBase = blockIdx.y * 128 + wave * 16;
  const bool hi = lane >= 16;
  const int bn = lane & 15;
  v8f acc[4] = {};
  gemm_strip(X, rowBase, W + (size_t)n0 * 512, 512, Bs, acc);
  const int bidx = rowBase >> 7;                        // batch
  const int k0 = (rowBase & (WINP - 1)) + (hi ? 8 : 0); // window pos of reg j=0
#pragma unroll
  for (int t = 0; t < 4; ++t) {
    int n = n0 + t * 16 + bn;
    float bv = bias[n];
    AF tmp;
#pragma unroll
    for (int j = 0; j < 8; ++j) tmp.h[j] = (_Float16)(acc[t][j] + bv);
    *(uint4*)(&YT[((size_t)bidx * DM + n) * WINP + k0]) = tmp.u[0];
  }
}

// ---------------- fused per-(b,h) attention ----------------
// K and V^T slabs are DMA'd to LDS once per block (shared by all 4 waves).
__global__ __launch_bounds__(128) void k_attn(const _Float16* __restrict__ Qh,
                                              const _Float16* __restrict__ Kh,
                                              const _Float16* __restrict__ VhT,
                                              float* __restrict__ att_out,
                                              _Float16* __restrict__ resh) {
  __shared__ alignas(16) _Float16 Ks[WINP * DK_];        // [win][dk] 16KB
  __shared__ alignas(16) _Float16 Vs[DK_ * WINP];        // [dk][win] 16KB
  __shared__ alignas(16) _Float16 att_lds[4][16 * WINP]; // 16KB
  const int b = blockIdx.x >> 3;
  const int h = blockIdx.x & 7;
  const int tid = threadIdx.x;
  const int lane = tid & 31, wave = tid >> 5;
  const bool hi = lane >= 16;
  const int am = lane & 15, akb = hi ? 8 : 0;
  const int bn = lane & 15, bkb = hi ? 8 : 0;

  const _Float16* Qp  = Qh  + ((size_t)b * TQ) * DM + h * DK_;
  const _Float16* Kp  = Kh  + ((size_t)b * WINP) * DM + h * DK_;
  const _Float16* Vtp = VhT + ((size_t)b * DM + h * DK_) * WINP;

  // async-stage K slab (1024 x 16B) then V slab (1024 x 16B); 8+8 per thread
#pragma unroll
  for (int s = 0; s < 8; ++s) {
    int seg = tid + s * 128;
    int n = seg >> 3, part = (seg & 7) * 8;
    async_cp16((uint32_t)(uintptr_t)(&Ks[n * DK_ + part]), Kp + (size_t)n * DM + part);
  }
#pragma unroll
  for (int s = 0; s < 8; ++s) {
    int seg = tid + s * 128;
    int d = seg >> 4, part = (seg & 15) * 8;
    async_cp16((uint32_t)(uintptr_t)(&Vs[d * WINP + part]), Vtp + (size_t)d * WINP + part);
  }
  asm volatile("s_wait_asynccnt 0x8" ::: "memory");   // K slab done (in-order), V in flight
  __syncthreads();

  // scores: 16(row strip) x 128(window) = 8 tiles, K = DK = 64
  v8f acc[8] = {};
#pragma unroll
  for (int kc = 0; kc < DK_; kc += 32) {
    AF a;
    const _Float16* ap = Qp + (size_t)(wave * 16 + am) * DM + kc + akb;
    a.u[0] = *(const uint4*)(ap);
    a.u[1] = *(const uint4*)(ap + 16);
#pragma unroll
    for (int t = 0; t < 8; ++t) {
      AF bf;
      const _Float16* bp = &Ks[(t * 16 + bn) * DK_ + kc + bkb];
      bf.u[0] = *(const uint4*)(bp);          // ds_load_b128
      bf.u[1] = *(const uint4*)(bp + 16);
      acc[t] = wmma16(a, bf, acc[t]);
    }
  }
  // softmax: row m lives in one 16-lane half (reg j, half = hi)
#pragma unroll
  for (int j = 0; j < 8; ++j) {
    float mx = -3.0e38f;
#pragma unroll
    for (int t = 0; t < 8; ++t) {
      float s = acc[t][j] * 0.125f;           // 1/sqrt(64)
      int col = t * 16 + (lane & 15);
      if (col >= WIN_) s = -1.0e30f;          // mask pad cols
      acc[t][j] = s;
      mx = fmaxf(mx, s);
    }
#pragma unroll
    for (int off = 1; off < 16; off <<= 1) mx = fmaxf(mx, __shfl_xor(mx, off, 32));
    float sm = 0.f;
#pragma unroll
    for (int t = 0; t < 8; ++t) { float e = __expf(acc[t][j] - mx); acc[t][j] = e; sm += e; }
#pragma unroll
    for (int off = 1; off < 16; off <<= 1) sm += __shfl_xor(sm, off, 32);
    float inv = 1.0f / sm;
#pragma unroll
    for (int t = 0; t < 8; ++t) acc[t][j] *= inv;
  }
  // emit att (f32 global) + f16 strip to LDS for A-fragment reload
#pragma unroll
  for (int t = 0; t < 8; ++t) {
#pragma unroll
    for (int j = 0; j < 8; ++j) {
      int m = j + (hi ? 8 : 0);
      int col = t * 16 + (lane & 15);
      float v = acc[t][j];
      att_lds[wave][m * WINP + col] = (_Float16)v;
      if (col < WIN_) {
        size_t row = (size_t)(h * B_ + b) * TQ + wave * 16 + m;
        att_out[row * WIN_ + col] = v;
      }
    }
  }
  asm volatile("s_wait_asynccnt 0x0" ::: "memory");   // V slab done
  __syncthreads();

  // res strip = att(16x128) @ V(128x64); both operands from LDS
  v8f racc[4] = {};
#pragma unroll
  for (int kc = 0; kc < WINP; kc += 32) {
    AF a;
    const _Float16* ap = &att_lds[wave][am * WINP + kc + akb];
    a.u[0] = *(const uint4*)(ap);
    a.u[1] = *(const uint4*)(ap + 16);
#pragma unroll
    for (int t = 0; t < 4; ++t) {
      AF bf;
      const _Float16* bp = &Vs[(t * 16 + bn) * WINP + kc + bkb];
      bf.u[0] = *(const uint4*)(bp);
      bf.u[1] = *(const uint4*)(bp + 16);
      racc[t] = wmma16(a, bf, racc[t]);
    }
  }
#pragma unroll
  for (int t = 0; t < 4; ++t) {
#pragma unroll
    for (int j = 0; j < 8; ++j) {
      int m = j + (hi ? 8 : 0);
      size_t row = (size_t)b * TQ + wave * 16 + m;
      resh[row * DM + h * DK_ + t * 16 + (lane & 15)] = (_Float16)racc[t][j];
    }
  }
}

// ---------------- out = [query|res] @ Wp^T + bp  (two K=512 passes) ----------------
__global__ __launch_bounds__(256) void k_gemm_out(const _Float16* __restrict__ Xq,
                                                  const _Float16* __restrict__ Xr,
                                                  const _Float16* __restrict__ Wph,
                                                  const float* __restrict__ bp,
                                                  float* __restrict__ out) {
  __shared__ alignas(16) _Float16 Bs[2][64 * 32];
  const int lane = threadIdx.x & 31, wave = threadIdx.x >> 5;
  const int n0 = blockIdx.x * 64;
  const int rowBase = blockIdx.y * 128 + wave * 16;
  const bool hi = lane >= 16;
  const int bn = lane & 15;
  v8f acc[4] = {};
  gemm_strip(Xq, rowBase, Wph + (size_t)n0 * 1024,       1024, Bs, acc);
  gemm_strip(Xr, rowBase, Wph + (size_t)n0 * 1024 + 512, 1024, Bs, acc);
#pragma unroll
  for (int t = 0; t < 4; ++t) {
    int n = n0 + t * 16 + bn;
    float bv = bp[n];
#pragma unroll
    for (int j = 0; j < 8; ++j) {
      int m = rowBase + j + (hi ? 8 : 0);
      out[(size_t)m * 512 + n] = acc[t][j] + bv;
    }
  }
}

// ---------------- new_center ----------------
__global__ __launch_bounds__(128) void k_center(const float* __restrict__ att,
                                                const int* __restrict__ center,
                                                float* __restrict__ outc) {
  __shared__ float red[128];
  int b = blockIdx.x;
  int start = center[b] - 48; if (start < 0) start = 0;
  float part = 0.f;
  for (int k = threadIdx.x; k < WIN_; k += blockDim.x) {
    float s = 0.f;
#pragma unroll
    for (int hh = 0; hh < H_; ++hh)
      s += att[((size_t)(hh * B_ + b) * TQ + (TQ - 1)) * WIN_ + k];
    part += (float)(start + k) * s;   // reference uses unclamped rng
  }
  red[threadIdx.x] = part;
  __syncthreads();
  for (int st = 64; st > 0; st >>= 1) {
    if ((int)threadIdx.x < st) red[threadIdx.x] += red[threadIdx.x + st];
    __syncthreads();
  }
  if (threadIdx.x == 0) outc[b] = floorf(red[0] / (float)H_);
}

extern "C" void kernel_launch(void* const* d_in, const int* in_sizes, int n_in,
                              void* d_out, int out_size, void* d_ws, size_t ws_size,
                              hipStream_t stream) {
  (void)in_sizes; (void)n_in; (void)out_size; (void)ws_size;
  const float* query  = (const float*)d_in[0];
  const float* values = (const float*)d_in[1];
  const int*   center = (const int*)  d_in[2];
  const float* Wk = (const float*)d_in[3];
  const float* bk = (const float*)d_in[4];
  const float* Wq = (const float*)d_in[5];
  const float* bq = (const float*)d_in[6];
  const float* Wv = (const float*)d_in[7];
  const float* bv = (const float*)d_in[8];
  const float* Wp = (const float*)d_in[9];
  const float* bp = (const float*)d_in[10];

  char* ws = (char*)d_ws;
  size_t off = 0;
  auto carve = [&](size_t halves) {
    _Float16* p = (_Float16*)(ws + off);
    off += ((halves * sizeof(_Float16) + 255) / 256) * 256;
    return p;
  };
  const size_t NQ = (size_t)B_ * TQ * DM;      // 2048*512
  const size_t NW = (size_t)B_ * WINP * DM;    // 4096*512
  _Float16* qh   = carve(NQ);
  _Float16* Wqh  = carve(512 * 512);
  _Float16* Wkh  = carve(512 * 512);
  _Float16* Wvh  = carve(512 * 512);
  _Float16* Wph  = carve(512 * 1024);
  _Float16* winh = carve(NW);
  _Float16* Qh   = carve(NQ);
  _Float16* Kh   = carve(NW);
  _Float16* VhT  = carve(NW);                  // transposed: [B][DM][WINP]
  _Float16* resh = carve(NQ);

  const size_t ATT_N = (size_t)H_ * B_ * TQ * WIN_;   // 1,589,248
  const size_t OUT_N = (size_t)B_ * TQ * DM;          // 1,048,576
  float* att_out = (float*)d_out;
  float* out_p   = att_out + ATT_N;
  float* cen_p   = out_p + OUT_N;

  auto G = [](size_t n) { return (unsigned)((n + 255) / 256); };

  const size_t PREP_N = NQ + 3 * (512 * 512) + 512 * 1024;
  k_prep<<<G(PREP_N), 256, 0, stream>>>(query, Wq, Wk, Wv, Wp, qh, Wqh, Wkh, Wvh, Wph);
  k_gather<<<G(NW), 256, 0, stream>>>(values, center, winh);

  k_gemm_proj <<<dim3(8, 16), 256, 0, stream>>>(qh,   Wqh, bq, Qh);
  k_gemm_proj <<<dim3(8, 32), 256, 0, stream>>>(winh, Wkh, bk, Kh);
  k_gemm_projT<<<dim3(8, 32), 256, 0, stream>>>(winh, Wvh, bv, VhT);

  k_attn<<<B_ * H_, 128, 0, stream>>>(Qh, Kh, VhT, att_out, resh);

  k_gemm_out<<<dim3(8, 16), 256, 0, stream>>>(qh, resh, Wph, bp, out_p);
  k_center<<<B_, 128, 0, stream>>>(att_out, center, cen_p);
}
